// DualAdjacencyFusion_49649821942402
// MI455X (gfx1250) — matrix operand order repaired and checked by reading, not compiled, not run on hardware
//
#include <hip/hip_runtime.h>

// ---------------------------------------------------------------------------
// Types for CDNA5 WMMA (wave32, 16x16x32 bf16 -> f32 accum)
// ---------------------------------------------------------------------------
typedef __attribute__((ext_vector_type(16))) __bf16 v16bf;
typedef __attribute__((ext_vector_type(8)))  __bf16 v8bf;
typedef __attribute__((ext_vector_type(8)))  float  v8f;

#define N_DIM   4096
#define D_DIM   512
#define V_DIM   3
#define TILE    128
#define KCH     32
#define LDSTR   40            // bf16 elements per LDS row (32 + pad, 16B aligned)
#define L_TH    0.8f
#define EPSF    1e-10f

#define BUF_BYTES (TILE * LDSTR * 2)   // 10240 B per LDS buffer

// round-to-nearest-even f32 -> bf16 (stored as ushort)
__device__ __forceinline__ unsigned short f2bf(float f) {
    unsigned int u = __float_as_uint(f);
    u += 0x7FFFu + ((u >> 16) & 1u);
    return (unsigned short)(u >> 16);
}

// ---------------------------------------------------------------------------
// CDNA5 async copy: 64 B (4 x b128) global -> LDS, tracked by ASYNCcnt.
// INST_OFFSET applies to both the LDS and global side (ISA 15.18.3 op 98).
// ---------------------------------------------------------------------------
__device__ __forceinline__ void async_copy64(unsigned ldsoff, const void* gsrc) {
    unsigned long long ga = (unsigned long long)(uintptr_t)gsrc;
    asm volatile(
        "global_load_async_to_lds_b128 %0, %1, off\n\t"
        "global_load_async_to_lds_b128 %0, %1, off offset:16\n\t"
        "global_load_async_to_lds_b128 %0, %1, off offset:32\n\t"
        "global_load_async_to_lds_b128 %0, %1, off offset:48"
        :
        : "v"(ldsoff), "v"(ga)
        : "memory");
}

__device__ __forceinline__ void wait_async0() {
#if __has_builtin(__builtin_amdgcn_s_wait_asynccnt)
    __builtin_amdgcn_s_wait_asynccnt(0);
#else
    asm volatile("s_wait_asynccnt 0" ::: "memory");
#endif
}

// ---------------------------------------------------------------------------
// Kernel: zero beta + per-block partials
// ---------------------------------------------------------------------------
__global__ __launch_bounds__(256) void init_zero(float* beta, float* part) {
    int t = blockIdx.x * 256 + threadIdx.x;
    if (t < 4) beta[t] = 0.0f;
    if (t < V_DIM * 1024) part[t] = 0.0f;
}

// ---------------------------------------------------------------------------
// Kernel: row-normalize A_v [3*4096 rows x 4096] f32 -> bf16
// ---------------------------------------------------------------------------
__global__ __launch_bounds__(256) void norm_rows_A(const float* __restrict__ A,
                                                   unsigned short* __restrict__ An) {
    const size_t row = blockIdx.x;
    const float4* src = (const float4*)(A + row * (size_t)N_DIM);
    ushort4* dst = (ushort4*)(An + row * (size_t)N_DIM);
    const int t = threadIdx.x;

    float4 x[4];
    float ss = 0.0f;
#pragma unroll
    for (int i = 0; i < 4; ++i) {
        x[i] = src[t + 256 * i];
        ss += x[i].x * x[i].x + x[i].y * x[i].y + x[i].z * x[i].z + x[i].w * x[i].w;
    }
#pragma unroll
    for (int off = 16; off >= 1; off >>= 1) ss += __shfl_down(ss, off, 32);

    __shared__ float red[8];
    __shared__ float scale_s;
    if ((t & 31) == 0) red[t >> 5] = ss;
    __syncthreads();
    if (t == 0) {
        float s = 0.0f;
#pragma unroll
        for (int i = 0; i < 8; ++i) s += red[i];
        scale_s = (s > 0.0f) ? rsqrtf(s) : 0.0f;
    }
    __syncthreads();
    const float sc = scale_s;
#pragma unroll
    for (int i = 0; i < 4; ++i) {
        ushort4 o;
        o.x = f2bf(x[i].x * sc); o.y = f2bf(x[i].y * sc);
        o.z = f2bf(x[i].z * sc); o.w = f2bf(x[i].w * sc);
        dst[t + 256 * i] = o;
    }
}

// ---------------------------------------------------------------------------
// Kernel: row-normalize feature [3*4096 rows x 512] f32 -> bf16
// ---------------------------------------------------------------------------
__global__ __launch_bounds__(128) void norm_rows_F(const float* __restrict__ F,
                                                   unsigned short* __restrict__ Fn) {
    const size_t row = blockIdx.x;
    const float4* src = (const float4*)(F + row * (size_t)D_DIM);
    ushort4* dst = (ushort4*)(Fn + row * (size_t)D_DIM);
    const int t = threadIdx.x;

    float4 x = src[t];
    float ss = x.x * x.x + x.y * x.y + x.z * x.z + x.w * x.w;
#pragma unroll
    for (int off = 16; off >= 1; off >>= 1) ss += __shfl_down(ss, off, 32);

    __shared__ float red[4];
    __shared__ float scale_s;
    if ((t & 31) == 0) red[t >> 5] = ss;
    __syncthreads();
    if (t == 0) {
        float s = red[0] + red[1] + red[2] + red[3];
        scale_s = (s > 0.0f) ? rsqrtf(s) : 0.0f;
    }
    __syncthreads();
    const float sc = scale_s;
    ushort4 o;
    o.x = f2bf(x.x * sc); o.y = f2bf(x.y * sc);
    o.z = f2bf(x.z * sc); o.w = f2bf(x.w * sc);
    dst[t] = o;
}

// ---------------------------------------------------------------------------
// LDS fragment loaders following the CDNA5 16-bit WMMA VGPR layouts.
// A 16x32 bf16: lanes 0-15 hold K 0..7 & 16..23; lanes 16-31 K 8..15 & 24..31
// B 32x16 bf16: lane = column; lanes 0-15 hold K 0..7 & 16..23; lanes 16-31 K 8..15 & 24..31
// ---------------------------------------------------------------------------
__device__ __forceinline__ v16bf load_frag_A(const unsigned short* lds, int rowBase, int lane) {
    const int r  = rowBase + (lane & 15);
    const int kh = (lane >> 4) * 8;
    v8bf a0 = *(const v8bf*)(lds + r * LDSTR + kh);
    v8bf a1 = *(const v8bf*)(lds + r * LDSTR + 16 + kh);
    return __builtin_shufflevector(a0, a1, 0, 1, 2, 3, 4, 5, 6, 7, 8, 9, 10, 11, 12, 13, 14, 15);
}

__device__ __forceinline__ v16bf load_frag_B(const unsigned short* lds, int colBase, int lane) {
    const int c  = colBase + (lane & 15);
    const int kh = (lane >> 4) * 8;
    v8bf b0 = *(const v8bf*)(lds + c * LDSTR + kh);
    v8bf b1 = *(const v8bf*)(lds + c * LDSTR + 16 + kh);
    return __builtin_shufflevector(b0, b1, 0, 1, 2, 3, 4, 5, 6, 7, 8, 9, 10, 11, 12, 13, 14, 15);
}

// ---------------------------------------------------------------------------
// One GEMM phase: acc(64x64 per wave) += Xi[128 x K] * Xj[128 x K]^T
// Software-pipelined: async global->LDS copy of chunk i+1 (double-buffered)
// overlaps the 16 WMMAs of chunk i.
// ---------------------------------------------------------------------------
__device__ __forceinline__ void gemm_phase(const unsigned short* __restrict__ Xi,
                                           const unsigned short* __restrict__ Xj,
                                           int K,
                                           unsigned short* ldsA,   // [2][TILE*LDSTR]
                                           unsigned short* ldsB,   // [2][TILE*LDSTR]
                                           v8f (&acc)[4][4]) {
    const int t    = threadIdx.x;
    const int lane = t & 31;
    const int wave = t >> 5;
    const int wr   = (wave >> 1) * 64;
    const int wc   = (wave & 1) * 64;
    const int nch  = K / KCH;

    const unsigned ldsA0 = (unsigned)(uintptr_t)(ldsA + t * LDSTR);
    const unsigned ldsB0 = (unsigned)(uintptr_t)(ldsB + t * LDSTR);
    const unsigned short* rowI = Xi + (size_t)t * K;
    const unsigned short* rowJ = Xj + (size_t)t * K;

    __syncthreads();                       // WAR: prior phase readers done before overwrite
    async_copy64(ldsA0, rowI);             // chunk 0 -> buffer 0
    async_copy64(ldsB0, rowJ);

    for (int i = 0; i < nch; ++i) {
        const int b = i & 1;
        wait_async0();                     // my chunk-i copies complete (in-order)
        __syncthreads();                   // all waves' copies landed; all readers of 1-b done

        if (i + 1 < nch) {                 // prefetch chunk i+1 into the other buffer
            const unsigned dst = (unsigned)((1 - b) * BUF_BYTES);
            async_copy64(ldsA0 + dst, rowI + (size_t)(i + 1) * KCH);
            async_copy64(ldsB0 + dst, rowJ + (size_t)(i + 1) * KCH);
        }

        const unsigned short* bufA = ldsA + b * (TILE * LDSTR);
        const unsigned short* bufB = ldsB + b * (TILE * LDSTR);
        v16bf af[4], bfr[4];
#pragma unroll
        for (int x = 0; x < 4; ++x) {
            af[x]  = load_frag_A(bufA, wr + x * 16, lane);
            bfr[x] = load_frag_B(bufB, wc + x * 16, lane);
        }
#pragma unroll
        for (int x = 0; x < 4; ++x)
#pragma unroll
            for (int y = 0; y < 4; ++y)
                acc[x][y] = __builtin_amdgcn_wmma_f32_16x16x32_bf16(
                    false, af[x], false, bfr[y], (short)0, acc[x][y], false, false);
    }
}

// ---------------------------------------------------------------------------
// Fused kernel: per 128x128 tile of (i,j):
//   S_feat tile (WMMA over K=512) -> pack l bits
//   S_v tile    (WMMA over K=4096) -> masked BCE -> partial[v][block]
// Exploits symmetry: skip bj<bi, double off-diagonal tiles.
// ---------------------------------------------------------------------------
__global__ __launch_bounds__(128) void fused_sim_bce(const unsigned short* __restrict__ An,
                                                     const unsigned short* __restrict__ Fn,
                                                     float* __restrict__ part) {
    __shared__ unsigned short ldsA[2 * TILE * LDSTR];
    __shared__ unsigned short ldsB[2 * TILE * LDSTR];
    __shared__ float red[4];

    const int v  = blockIdx.z;
    const int bx = blockIdx.x, by = blockIdx.y;
    const int bi = bx * TILE, bj = by * TILE;
    const int t = threadIdx.x, lane = t & 31, wave = t >> 5;
    float* slot = part + v * 1024 + by * 32 + bx;

    if (bj < bi) {               // symmetric half skipped; keep partial deterministic
        if (t == 0) *slot = 0.0f;
        return;
    }

    v8f acc[4][4];
#pragma unroll
    for (int x = 0; x < 4; ++x)
#pragma unroll
        for (int y = 0; y < 4; ++y)
            acc[x][y] = (v8f){0.f, 0.f, 0.f, 0.f, 0.f, 0.f, 0.f, 0.f};

    // ---- phase A: S_feat tile, K = 512 ----
    const unsigned short* Fi = Fn + ((size_t)v * N_DIM + bi) * D_DIM;
    const unsigned short* Fj = Fn + ((size_t)v * N_DIM + bj) * D_DIM;
    gemm_phase(Fi, Fj, D_DIM, ldsA, ldsB, acc);

    unsigned int lm[4] = {0u, 0u, 0u, 0u};
#pragma unroll
    for (int x = 0; x < 4; ++x)
#pragma unroll
        for (int y = 0; y < 4; ++y)
#pragma unroll
            for (int e = 0; e < 8; ++e) {
                const int idx = (x * 4 + y) * 8 + e;
                if (acc[x][y][e] > L_TH) lm[idx >> 5] |= (1u << (idx & 31));
            }

#pragma unroll
    for (int x = 0; x < 4; ++x)
#pragma unroll
        for (int y = 0; y < 4; ++y)
            acc[x][y] = (v8f){0.f, 0.f, 0.f, 0.f, 0.f, 0.f, 0.f, 0.f};

    // ---- phase B: S_v tile, K = 4096 ----
    const unsigned short* Ai = An + ((size_t)v * N_DIM + bi) * (size_t)N_DIM;
    const unsigned short* Aj = An + ((size_t)v * N_DIM + bj) * (size_t)N_DIM;
    gemm_phase(Ai, Aj, N_DIM, ldsA, ldsB, acc);

    // ---- masked BCE epilogue ----
    float lsum = 0.0f;
#pragma unroll
    for (int x = 0; x < 4; ++x)
#pragma unroll
        for (int y = 0; y < 4; ++y)
#pragma unroll
            for (int e = 0; e < 8; ++e) {
                const float s = acc[x][y][e];
                if (s > EPSF && (1.0f - s) > EPSF) {
                    const int idx = (x * 4 + y) * 8 + e;
                    const bool lb = (lm[idx >> 5] >> (idx & 31)) & 1u;
                    lsum -= __logf(lb ? s : (1.0f - s));
                }
            }
    if (bi != bj) lsum *= 2.0f;   // symmetric twin tile

#pragma unroll
    for (int off = 16; off >= 1; off >>= 1) lsum += __shfl_down(lsum, off, 32);
    if (lane == 0) red[wave] = lsum;
    __syncthreads();
    if (t == 0) *slot = red[0] + red[1] + red[2] + red[3];
}

// ---------------------------------------------------------------------------
// Deterministic reduction of 1024 partials per view -> beta[v]
// ---------------------------------------------------------------------------
__global__ __launch_bounds__(256) void reduce_beta(const float* __restrict__ part,
                                                   float* __restrict__ beta) {
    const int v = blockIdx.x, t = threadIdx.x;
    float s = part[v * 1024 + t] + part[v * 1024 + 256 + t] +
              part[v * 1024 + 512 + t] + part[v * 1024 + 768 + t];
#pragma unroll
    for (int off = 16; off >= 1; off >>= 1) s += __shfl_down(s, off, 32);
    __shared__ float red[8];
    if ((t & 31) == 0) red[t >> 5] = s;
    __syncthreads();
    if (t == 0) {
        float b = 0.0f;
#pragma unroll
        for (int i = 0; i < 8; ++i) b += red[i];
        beta[v] = b;
    }
}

// ---------------------------------------------------------------------------
// w = softmax(min(beta, 100))
// ---------------------------------------------------------------------------
__global__ void softmax3(const float* __restrict__ beta, float* __restrict__ w) {
    if (threadIdx.x == 0) {
        float b0 = fminf(beta[0], 100.0f);
        float b1 = fminf(beta[1], 100.0f);
        float b2 = fminf(beta[2], 100.0f);
        float m  = fmaxf(b0, fmaxf(b1, b2));
        float e0 = __expf(b0 - m), e1 = __expf(b1 - m), e2 = __expf(b2 - m);
        float inv = 1.0f / (e0 + e1 + e2);
        w[0] = e0 * inv; w[1] = e1 * inv; w[2] = e2 * inv;
    }
}

// ---------------------------------------------------------------------------
// A_c = sum_v w[v] * A_v   (pure bandwidth, float4)
// ---------------------------------------------------------------------------
__global__ __launch_bounds__(256) void weighted_sum(const float* __restrict__ A,
                                                    const float* __restrict__ w,
                                                    float* __restrict__ out) {
    const size_t i = (size_t)blockIdx.x * 256 + threadIdx.x;
    const size_t S = (size_t)N_DIM * N_DIM / 4;
    const float w0 = w[0], w1 = w[1], w2 = w[2];
    const float4* a = (const float4*)A;
    float4 x = a[i], y = a[i + S], z = a[i + 2 * S];
    float4 o;
    o.x = w0 * x.x + w1 * y.x + w2 * z.x;
    o.y = w0 * x.y + w1 * y.y + w2 * z.y;
    o.z = w0 * x.z + w1 * y.z + w2 * z.z;
    o.w = w0 * x.w + w1 * y.w + w2 * z.w;
    ((float4*)out)[i] = o;
}

// ---------------------------------------------------------------------------
// Launch
// ---------------------------------------------------------------------------
extern "C" void kernel_launch(void* const* d_in, const int* in_sizes, int n_in,
                              void* d_out, int out_size, void* d_ws, size_t ws_size,
                              hipStream_t stream) {
    (void)in_sizes; (void)n_in; (void)out_size; (void)ws_size;
    const float* A_v  = (const float*)d_in[0];   // [3,4096,4096] f32
    const float* feat = (const float*)d_in[1];   // [3,4096,512]  f32
    float* out = (float*)d_out;                  // [4096,4096]   f32

    char* ws = (char*)d_ws;
    const size_t OFF_FN   = (size_t)V_DIM * N_DIM * N_DIM * 2;      // 100,663,296
    const size_t OFF_BETA = OFF_FN + (size_t)V_DIM * N_DIM * D_DIM * 2;
    const size_t OFF_W    = OFF_BETA + 64;
    const size_t OFF_PART = OFF_W + 64;

    unsigned short* An   = (unsigned short*)ws;
    unsigned short* Fn   = (unsigned short*)(ws + OFF_FN);
    float*          beta = (float*)(ws + OFF_BETA);
    float*          w    = (float*)(ws + OFF_W);
    float*          part = (float*)(ws + OFF_PART);

    init_zero<<<(V_DIM * 1024 + 255) / 256, 256, 0, stream>>>(beta, part);
    norm_rows_A<<<V_DIM * N_DIM, 256, 0, stream>>>(A_v, An);
    norm_rows_F<<<V_DIM * N_DIM, 128, 0, stream>>>(feat, Fn);
    fused_sim_bce<<<dim3(N_DIM / TILE, N_DIM / TILE, V_DIM), 128, 0, stream>>>(An, Fn, part);
    reduce_beta<<<V_DIM, 256, 0, stream>>>(part, beta);
    softmax3<<<1, 32, 0, stream>>>(beta, w);
    weighted_sum<<<(N_DIM * (size_t)N_DIM / 4) / 256, 256, 0, stream>>>(A_v, w, out);
}
